// MultiHeadSelfAttention_59794534695249
// MI455X (gfx1250) — matrix operand order repaired and checked by reading, not compiled, not run on hardware
//
#include <hip/hip_runtime.h>
#include <hip/hip_bf16.h>
#include <cstddef>

// ---------------------------------------------------------------------------
// MI455X / gfx1250 multi-head self-attention (degenerate 8-identical-heads
// variant). All matmuls via v_wmma_f32_16x16x32_bf16; f32 accumulate;
// softmax in f32. Operands pre-converted to bf16 once (bandwidth-trivial)
// so the GEMM/attention inner loops are pure b128-load + WMMA.
// ---------------------------------------------------------------------------

typedef __attribute__((ext_vector_type(16))) __bf16 v16bf;
typedef __attribute__((ext_vector_type(8)))  __bf16 v8bf;
typedef __attribute__((ext_vector_type(8)))  float  v8f;

#define DHID   512
#define SEQ    4096
#define MROWS  8192          // B*S
#define BKEY   128           // key block for attention
#define SCALE_INV 0.125f     // 1/sqrt(512/8) = 1/8

// ---------------- WMMA fragment helpers (bf16 16x16x32; contiguous k order used by this code) -------
// A (16x32, MxK): lane L (M = L&15): elems 0..15 -> K = kb + (L>>4)*16 + e
// B (32x16, KxN): lane L (N = L&15): elems 0..15 -> K = kb + (L>>4)*16 + e
// C/D (16x16 f32): lane L, vgpr v: row M = v + 8*(L>>4), col N = L&15

__device__ __forceinline__ v8f wmma_bf16(v16bf a, v16bf b, v8f c) {
  return __builtin_amdgcn_wmma_f32_16x16x32_bf16(
      /*neg_a=*/false, a, /*neg_b=*/false, b,
      /*c_mod=*/(short)0, c, /*reuse_a=*/false, /*reuse_b=*/false);
}

// A-fragment from a bf16 row (two 16B loads)
__device__ __forceinline__ v16bf frag_a(const __bf16* rowp, int kb, int lh) {
  v8bf a = *(const v8bf*)(rowp + kb + lh * 16);
  v8bf b = *(const v8bf*)(rowp + kb + lh * 16 + 8);
  v16bf r;
#pragma unroll
  for (int i = 0; i < 8; ++i) { r[i] = a[i]; r[8 + i] = b[i]; }
  return r;
}

// B-fragment: 16 contiguous bf16 (two 16B loads)
__device__ __forceinline__ v16bf frag_b(const __bf16* p) {
  v8bf a = ((const v8bf*)p)[0];
  v8bf b = ((const v8bf*)p)[1];
  v16bf r;
#pragma unroll
  for (int i = 0; i < 8; ++i) { r[i] = a[i]; r[8 + i] = b[i]; }
  return r;
}

// ---------------- fp32 -> bf16 pre-conversion (bandwidth-trivial) -----------
__global__ __launch_bounds__(256) void f32_to_bf16(const float* __restrict__ in,
                                                   __bf16* __restrict__ out, int n) {
  const int i = (blockIdx.x * 256 + threadIdx.x) * 8;
  if (i + 8 <= n) {
    float4 a = ((const float4*)(in + i))[0];
    float4 b = ((const float4*)(in + i))[1];
    v8bf r;
    r[0] = (__bf16)a.x; r[1] = (__bf16)a.y; r[2] = (__bf16)a.z; r[3] = (__bf16)a.w;
    r[4] = (__bf16)b.x; r[5] = (__bf16)b.y; r[6] = (__bf16)b.z; r[7] = (__bf16)b.w;
    *(v8bf*)(out + i) = r;
  }
}

// ---------------- GEMM: out[m,n] = scale * sum_k A[m,k]*W[n,k] + bias[n] ----
// M = 8192, N = K = 512, all-bf16 operands. WG = 256 threads (8 waves);
// wave w -> rows [bx*128 + w*16, +16), cols [by*64, +64) (4 accum tiles).
template <bool TRANS_STORE, bool OUT_F32>
__global__ __launch_bounds__(256) void gemm512(const __bf16* __restrict__ A,
                                               const __bf16* __restrict__ W,
                                               const float* __restrict__ bias,
                                               void* __restrict__ outv,
                                               float scale) {
  const int lane = threadIdx.x & 31;
  const int wave = threadIdx.x >> 5;
  const int lr = lane & 15;
  const int lh = lane >> 4;
  const int m0 = blockIdx.x * 128 + wave * 16;
  const int n0 = blockIdx.y * 64;

  const __bf16* arow = A + (size_t)(m0 + lr) * DHID;

  v8f acc[4] = {};

#pragma unroll 4
  for (int ks = 0; ks < 16; ++ks) {
    const int kb = ks * 32;
    v16bf af = frag_a(arow, kb, lh);
#pragma unroll
    for (int t = 0; t < 4; ++t) {
      const __bf16* wrow = W + (size_t)(n0 + t * 16 + lr) * DHID + kb + lh * 16;
      acc[t] = wmma_bf16(af, frag_b(wrow), acc[t]);
    }
  }

#pragma unroll
  for (int t = 0; t < 4; ++t) {
    const int n = n0 + t * 16 + lr;
    const float bv = bias[n];
#pragma unroll
    for (int v = 0; v < 8; ++v) {
      const int m = m0 + v + 8 * lh;
      const float val = acc[t][v] * scale + bv;
      if constexpr (OUT_F32) {
        ((float*)outv)[(size_t)m * DHID + n] = val;
      } else if constexpr (TRANS_STORE) {
        // store v transposed per batch: vT[b][h][s]
        const int bb = m >> 12, s = m & 4095;
        ((__bf16*)outv)[((size_t)(bb * DHID + n)) * SEQ + s] = (__bf16)val;
      } else {
        ((__bf16*)outv)[(size_t)m * DHID + n] = (__bf16)val;
      }
    }
  }
}

// ---------------- Flash attention (mask: allowed j >= i) --------------------
// WG = 256 threads (8 waves) owns 16 query rows. Key blocks of 128, starting
// at floor(i0/128)*128 (reversed-causal). Wave w owns O columns [w*64, +64).
__global__ __launch_bounds__(256) void attn_kernel(const __bf16* __restrict__ qg,
                                                   const __bf16* __restrict__ kg,
                                                   const __bf16* __restrict__ vtg,
                                                   __bf16* __restrict__ og) {
  __shared__ __align__(16) float  Sblk[16][132];   // padded: conflict-free rows
  __shared__ __align__(16) __bf16 Pblk[16][136];   // 136*2B = 272B row stride
  __shared__ float red[16][17];                    // row partial max/sum
  __shared__ float row_m[16], row_l[16], row_f[16];

  const int lane = threadIdx.x & 31;
  const int wave = threadIdx.x >> 5;
  const int lr = lane & 15;
  const int lh = lane >> 4;
  const int b  = blockIdx.x >> 8;          // batch
  const int i0 = (blockIdx.x & 255) << 4;  // query tile start

  // softmax work split: 16 threads per row, 8 columns per thread
  const int srow = threadIdx.x >> 4;       // 0..15
  const int scg  = threadIdx.x & 15;       // column group, cols [scg*8, +8)

  const __bf16* q  = qg  + (size_t)b * SEQ * DHID;
  const __bf16* k  = kg  + (size_t)b * SEQ * DHID;
  const __bf16* vt = vtg + (size_t)b * DHID * SEQ;
  __bf16*       o  = og  + (size_t)b * SEQ * DHID;

  if (threadIdx.x < 16) { row_m[threadIdx.x] = -__builtin_inff(); row_l[threadIdx.x] = 0.f; }

  // Preload the 16x512 Q tile as 16 A-fragments (fully unrolled => stays in VGPRs)
  v16bf qf[16];
  const __bf16* qrow = q + (size_t)(i0 + lr) * DHID;
#pragma unroll
  for (int ks = 0; ks < 16; ++ks) qf[ks] = frag_a(qrow, ks * 32, lh);

  v8f oacc[4] = {};
  const int h0 = wave * 64;
  __syncthreads();

  for (int j0 = (i0 >> 7) << 7; j0 < SEQ; j0 += BKEY) {
    // ---- Phase 1: S sub-tile (16 x 16), columns [wave*16, +16) ----
    v8f st = {};
    const __bf16* krow = k + (size_t)(j0 + wave * 16 + lr) * DHID;
    if (j0 + BKEY < SEQ)   // prefetch next key block's rows for this wave
      __builtin_prefetch(krow + (size_t)BKEY * DHID, 0, 3);
#pragma unroll
    for (int ks = 0; ks < 16; ++ks) {
      st = wmma_bf16(qf[ks], frag_b(krow + ks * 32 + lh * 16), st);
    }
#pragma unroll
    for (int v = 0; v < 8; ++v) {
      const int r = v + 8 * lh;
      const int c = wave * 16 + lr;
      float s = st[v] * SCALE_INV;
      if (j0 + c < i0 + r) s = -__builtin_inff();   // mask: only j >= i allowed
      Sblk[r][c] = s;
    }
    __syncthreads();

    // ---- Phase 2a: per-thread partial max (16 thr/row x 8 cols) ----
    {
      float pm = -__builtin_inff();
#pragma unroll
      for (int c = 0; c < 8; ++c) pm = fmaxf(pm, Sblk[srow][scg * 8 + c]);
      red[srow][scg] = pm;
    }
    __syncthreads();

    // ---- Phase 2b: row max reduce + running-max update (16 threads) ----
    if (threadIdx.x < 16) {
      const int r = threadIdx.x;
      float mb = red[r][0];
#pragma unroll
      for (int g = 1; g < 16; ++g) mb = fmaxf(mb, red[r][g]);
      const float mo = row_m[r];
      const float mn = fmaxf(mo, mb);
      row_f[r] = __expf(mo - mn);            // 0 on first block (mo = -inf)
      row_m[r] = mn;
    }
    __syncthreads();

    // ---- Phase 2c: exp + P write (bf16) + partial sums ----
    {
      const float mn = row_m[srow];
      float ps = 0.f;
#pragma unroll
      for (int c = 0; c < 8; ++c) {
        const float p = __expf(Sblk[srow][scg * 8 + c] - mn);
        Pblk[srow][scg * 8 + c] = (__bf16)p;
        ps += p;
      }
      red[srow][scg] = ps;
    }
    __syncthreads();

    // ---- Phase 2d: row sum reduce (16 threads, overlaps with phase 3) ----
    if (threadIdx.x < 16) {
      const int r = threadIdx.x;
      float s = 0.f;
#pragma unroll
      for (int g = 0; g < 16; ++g) s += red[r][g];
      row_l[r] = row_l[r] * row_f[r] + s;
    }

    // ---- Phase 3: rescale O, accumulate P (16x128) x V (128x64) ----
    float fv[8];
#pragma unroll
    for (int v = 0; v < 8; ++v) fv[v] = row_f[v + 8 * lh];
#pragma unroll
    for (int t = 0; t < 4; ++t)
#pragma unroll
      for (int v = 0; v < 8; ++v) oacc[t][v] *= fv[v];

#pragma unroll
    for (int ks = 0; ks < 4; ++ks) {
      v16bf pa = frag_a(&Pblk[lr][0], ks * 32, lh);   // ds_load_b128 x2
#pragma unroll
      for (int t = 0; t < 4; ++t) {
        const __bf16* vrow = vt + (size_t)(h0 + t * 16 + lr) * SEQ + j0 + ks * 32 + lh * 16;
        oacc[t] = wmma_bf16(pa, frag_b(vrow), oacc[t]);
      }
    }
  }
  __syncthreads();   // make final row_l visible to all threads

  // ---- Finalize: O / l, store bf16 attn tile ----
  float linv[8];
#pragma unroll
  for (int v = 0; v < 8; ++v) linv[v] = 1.0f / row_l[v + 8 * lh];
#pragma unroll
  for (int t = 0; t < 4; ++t) {
    const int n = h0 + t * 16 + lr;
#pragma unroll
    for (int v = 0; v < 8; ++v) {
      const int m = i0 + v + 8 * lh;
      o[(size_t)m * DHID + n] = (__bf16)(oacc[t][v] * linv[v]);
    }
  }
}

// ---------------------------------------------------------------------------
extern "C" void kernel_launch(void* const* d_in, const int* in_sizes, int n_in,
                              void* d_out, int out_size, void* d_ws, size_t ws_size,
                              hipStream_t stream) {
  (void)in_sizes; (void)n_in; (void)out_size; (void)ws_size;
  const float* x  = (const float*)d_in[0];
  const float* Wq = (const float*)d_in[1];
  const float* bq = (const float*)d_in[2];
  const float* Wk = (const float*)d_in[3];
  const float* bk = (const float*)d_in[4];
  const float* Wv = (const float*)d_in[5];
  const float* bv = (const float*)d_in[6];
  const float* Wm = (const float*)d_in[7];
  const float* bm = (const float*)d_in[8];

  char* ws = (char*)d_ws;
  const size_t big = (size_t)MROWS * DHID * sizeof(__bf16);   // 8 MiB
  const size_t wsz = (size_t)DHID * DHID * sizeof(__bf16);    // 512 KiB
  __bf16* xb  = (__bf16*)(ws);
  __bf16* qb  = (__bf16*)(ws + 1 * big);
  __bf16* kb  = (__bf16*)(ws + 2 * big);
  __bf16* vtb = (__bf16*)(ws + 3 * big);   // transposed per batch: [b][h][s]
  __bf16* ab  = (__bf16*)(ws + 4 * big);
  __bf16* wqb = (__bf16*)(ws + 5 * big);
  __bf16* wkb = (__bf16*)(ws + 5 * big + 1 * wsz);
  __bf16* wvb = (__bf16*)(ws + 5 * big + 2 * wsz);
  __bf16* wmb = (__bf16*)(ws + 5 * big + 3 * wsz);

  dim3 tt(256);

  // bf16 pre-conversion: x (4M elems) + 4 weight matrices (256K each)
  f32_to_bf16<<<dim3(MROWS * DHID / 2048), tt, 0, stream>>>(x,  xb,  MROWS * DHID);
  f32_to_bf16<<<dim3(DHID * DHID / 2048), tt, 0, stream>>>(Wq, wqb, DHID * DHID);
  f32_to_bf16<<<dim3(DHID * DHID / 2048), tt, 0, stream>>>(Wk, wkb, DHID * DHID);
  f32_to_bf16<<<dim3(DHID * DHID / 2048), tt, 0, stream>>>(Wv, wvb, DHID * DHID);
  f32_to_bf16<<<dim3(DHID * DHID / 2048), tt, 0, stream>>>(Wm, wmb, DHID * DHID);

  dim3 gg(MROWS / 128, DHID / 64);
  gemm512<false, false><<<gg, tt, 0, stream>>>(xb, wqb, bq, qb, 1.0f);
  gemm512<false, false><<<gg, tt, 0, stream>>>(xb, wkb, bk, kb, 1.0f);
  gemm512<true,  false><<<gg, tt, 0, stream>>>(xb, wvb, bv, vtb, 1.0f);

  attn_kernel<<<dim3(2 * SEQ / 16), tt, 0, stream>>>(qb, kb, vtb, ab);

  // out = 8 * attn @ Wm^T + bm   (NUM_HEADS identical heads summed)
  gemm512<false, true><<<gg, tt, 0, stream>>>(ab, wmb, bm, d_out, 8.0f);
}
